// MultiHeadAttention_53721450938961
// MI455X (gfx1250) — matrix operand order repaired and checked by reading, not compiled, not run on hardware
//
#include <hip/hip_runtime.h>
#include <hip/hip_bf16.h>

// ---------------- problem constants ----------------
#define BATCH   4
#define SEQ     2048
#define EMBED   1024
#define NHEAD   16
#define HDIM    64
#define MROWS   (BATCH * SEQ)          // 8192
#define SCALE_F 0.125f                 // 1/sqrt(64)

typedef __attribute__((ext_vector_type(16))) _Float16 v16h;
typedef __attribute__((ext_vector_type(8)))  _Float16 v8h;
typedef __attribute__((ext_vector_type(4)))  _Float16 v4h;
typedef __attribute__((ext_vector_type(8)))  float    v8f;
typedef __attribute__((ext_vector_type(4)))  float    v4f;
typedef __attribute__((ext_vector_type(4)))  unsigned int v4u;
typedef __attribute__((ext_vector_type(8)))  int      v8i;
typedef __attribute__((ext_vector_type(4)))  int      v4i;

#if defined(__gfx1250__) && __has_builtin(__builtin_amdgcn_tensor_load_to_lds)
#define USE_TDM 1
#else
#define USE_TDM 0
#endif

// ---------------------------------------------------------------------------
// CDNA5 async global->LDS copy (ASYNCcnt) via inline asm; per-lane addressed.
// Low 32 bits of a generic pointer to LDS are the LDS byte address.
__device__ __forceinline__ void async_copy_b128(void* lds_dst, const void* gsrc) {
#if defined(__HIP_DEVICE_COMPILE__)
    unsigned int l = (unsigned int)(uintptr_t)lds_dst;
    asm volatile("global_load_async_to_lds_b128 %0, %1, off"
                 :: "v"(l), "v"(gsrc) : "memory");
#endif
}
__device__ __forceinline__ void wait_async0() {
#if defined(__HIP_DEVICE_COMPILE__)
    asm volatile("s_wait_asynccnt 0x0" ::: "memory");
#endif
}

// ---------------------------------------------------------------------------
// Tensor Data Mover: 2-D f16 tile load, global -> LDS, with LDS row padding.
// Descriptor packing per CDNA5 ISA ch.8 (group0 128b, group1 256b).
//   tile_x elems per row (f16), tile_y rows, row stride stride_x elems.
//   pad_interval_code/pad_amount_code encode LDS padding (DWORD units).
#if USE_TDM
__device__ __forceinline__ void tdm_load_2d_f16(
    unsigned int lds_addr, const void* gaddr,
    unsigned int tile_x, unsigned int tile_y, unsigned int stride_x,
    unsigned int pad_en, unsigned int pad_interval_code, unsigned int pad_amount_code)
{
    unsigned long long ga = (unsigned long long)(uintptr_t)gaddr;
    v4u g0;
    g0[0] = 1u;                                          // count=1, user desc
    g0[1] = lds_addr;                                    // lds_addr [63:32]
    g0[2] = (unsigned int)(ga & 0xFFFFFFFFu);            // global_addr lo
    g0[3] = (unsigned int)((ga >> 32) & 0x01FFFFFFu)     // global_addr [56:32]
          | (2u << 30);                                  // type=2 ("image")
    v8i g1;
    g1[0] = (int)((1u << 16)                             // data_size=1 -> 2B
                | (pad_en << 20)
                | (pad_interval_code << 22)
                | (pad_amount_code << 25));
    g1[1] = (int)((tile_x & 0xFFFFu) << 16);             // tensor_dim0 lo16 (=tile)
    g1[2] = (int)(((tile_x >> 16) & 0xFFFFu)             // tensor_dim0 hi16
                | ((tile_y & 0xFFFFu) << 16));           // tensor_dim1 lo16 (=tile)
    g1[3] = (int)(((tile_y >> 16) & 0xFFFFu)             // tensor_dim1 hi16
                | ((tile_x & 0xFFFFu) << 16));           // tile_dim0
    g1[4] = (int)(tile_y & 0xFFFFu);                     // tile_dim1 (tile_dim2=0)
    g1[5] = (int)stride_x;                               // tensor_dim0_stride lo32
    g1[6] = 0;                                           // stride0 hi / stride1 lo
    g1[7] = 0;                                           // stride1 hi
    v4i g2 = {0, 0, 0, 0};
    v4i g3 = {0, 0, 0, 0};
    v8i g4 = {0, 0, 0, 0, 0, 0, 0, 0};                   // 6-arg (clang-23) form
    __builtin_amdgcn_tensor_load_to_lds(g0, g1, g2, g3, g4, 0);
}
__device__ __forceinline__ void wait_tensor0() {
    __builtin_amdgcn_s_wait_tensorcnt((short)0);
}
#endif

// ---------------------------------------------------------------------------
// LDS fragment gathers (layouts per CDNA5 ISA 7.12.2, wave32)
// A-matrix 16x32 f16: lane<16 -> row=lane, K = {0..7, 16..23}
//                     lane>=16 -> row=lane-16, K = {8..15, 24..31}
__device__ __forceinline__ v16h frag_a(const _Float16* tile, int ld, int row_off, int k_off) {
    int lane = threadIdx.x & 31;
    int r  = row_off + (lane & 15);
    int k0 = k_off + ((lane >> 4) << 3);          // 0 or 8
    const _Float16* p = tile + r * ld + k0;
    v8h lo = *(const v8h*)(p);
    v8h hi = *(const v8h*)(p + 16);
    v16h f;
#pragma unroll
    for (int i = 0; i < 8; ++i) { f[i] = lo[i]; f[i + 8] = hi[i]; }
    return f;
}

// B-matrix 32x16 f16 (KxN): lane<16 -> col=lane, K = {0..7, 16..23}
//                           lane>=16 -> col=lane-16, K = {8..15, 24..31}
__device__ __forceinline__ v16h frag_b(const _Float16* tile, int ld, int col_off, int k_off) {
    int lane = threadIdx.x & 31;
    int c  = col_off + (lane & 15);
    int k0 = k_off + ((lane >> 4) << 3);          // 0 or 8
    const _Float16* p = tile + c * ld + k0;
    v8h lo = *(const v8h*)(p);
    v8h hi = *(const v8h*)(p + 16);
    v16h f;
#pragma unroll
    for (int i = 0; i < 8; ++i) { f[i] = lo[i]; f[i + 8] = hi[i]; }
    return f;
}

// ---------------------------------------------------------------------------
// C = A @ W^T + bias.  A:[M,K] (f32 or f16), W:[N,K] f32, C:[M,N] (f16 or f32)
// Block: 256 threads = 8 waves; BM=128, BN=64, BK=32; each wave owns 32x32.
template <typename AT, typename OT>
__global__ __launch_bounds__(256)
void gemm_bias_wmma(const AT* __restrict__ A, const float* __restrict__ W,
                    const float* __restrict__ bias, OT* __restrict__ C,
                    int M, int N, int K) {
    constexpr int BM = 128, BN = 64, BK = 32, LD = 40;   // LD*2B = 80B, 16B aligned
    __shared__ _Float16 As[BM * LD];
    __shared__ _Float16 Bs[BN * LD];

    const int t    = threadIdx.x;
    const int lane = t & 31;
    const int wave = t >> 5;
    const int wr   = wave >> 1;          // 0..3  (M dir)
    const int wc   = wave & 1;           // 0..1  (N dir)
    const int mbase = blockIdx.x * BM;
    const int nbase = blockIdx.y * BN;

    v8f acc[2][2] = {};

    for (int kb = 0; kb < K; kb += BK) {
        // prefetch next k-slab while staging this one
        if (kb + BK < K) {
            __builtin_prefetch(&A[(size_t)(mbase + (t & (BM - 1))) * K + kb + BK], 0, 0);
            __builtin_prefetch(&W[(size_t)(nbase + (t & (BN - 1))) * K + kb + BK], 0, 0);
        }
        // ---- stage A tile (128x32) ----
        if constexpr (sizeof(AT) == 4) {
#pragma unroll
            for (int i = 0; i < 4; ++i) {
                int idx = t + i * 256;                // 1024 chunks of 4 floats
                int row = idx >> 3, c4 = idx & 7;
                v4f v = *(const v4f*)(&A[(size_t)(mbase + row) * K + kb + c4 * 4]);
                v4h h;
#pragma unroll
                for (int j = 0; j < 4; ++j) h[j] = (_Float16)v[j];
                *(v4h*)(&As[row * LD + c4 * 4]) = h;
            }
        } else {
            // f16 A: CDNA5 async global->LDS copies (ASYNCcnt)
#pragma unroll
            for (int i = 0; i < 2; ++i) {
                int idx = t + i * 256;                // 512 chunks of 8 halves
                int row = idx >> 2, c8 = idx & 3;
                async_copy_b128(&As[row * LD + c8 * 8],
                                &A[(size_t)(mbase + row) * K + kb + c8 * 8]);
            }
        }
        // ---- stage W tile (64x32), W is [N,K] so row n is column n of B ----
#pragma unroll
        for (int i = 0; i < 2; ++i) {
            int idx = t + i * 256;                    // 512 chunks of 4 floats
            int row = idx >> 3, c4 = idx & 7;
            v4f v = *(const v4f*)(&W[(size_t)(nbase + row) * K + kb + c4 * 4]);
            v4h h;
#pragma unroll
            for (int j = 0; j < 4; ++j) h[j] = (_Float16)v[j];
            *(v4h*)(&Bs[row * LD + c4 * 4]) = h;
        }
        if constexpr (sizeof(AT) == 2) wait_async0();
        __syncthreads();

        v16h af[2], bf[2];
        af[0] = frag_a(As, LD, wr * 32, 0);
        af[1] = frag_a(As, LD, wr * 32 + 16, 0);
        bf[0] = frag_b(Bs, LD, wc * 32, 0);
        bf[1] = frag_b(Bs, LD, wc * 32 + 16, 0);
#pragma unroll
        for (int mf = 0; mf < 2; ++mf)
#pragma unroll
            for (int nf = 0; nf < 2; ++nf)
                acc[mf][nf] = __builtin_amdgcn_wmma_f32_16x16x32_f16(
                    false, af[mf], false, bf[nf], (short)0, acc[mf][nf], false, false);
        __syncthreads();
    }

    // ---- epilogue: D layout (lane<16: rows 0..7, lane>=16: rows 8..15) ----
    const int half = lane >> 4;
    const int col0 = lane & 15;
#pragma unroll
    for (int mf = 0; mf < 2; ++mf)
#pragma unroll
        for (int nf = 0; nf < 2; ++nf)
#pragma unroll
            for (int r = 0; r < 8; ++r) {
                int row = mbase + wr * 32 + mf * 16 + half * 8 + r;
                int col = nbase + wc * 32 + nf * 16 + col0;
                float v = acc[mf][nf][r] + bias[col];
                C[(size_t)row * N + col] = (OT)v;
            }
}

// ---------------------------------------------------------------------------
// Causal flash-attention tile kernel (two-pass softmax, stats in registers).
// Grid: (SEQ/64, NHEAD, BATCH). Block: 128 threads = 4 waves x 16 q-rows.
// Q/K tiles staged by the Tensor Data Mover (TENSORcnt) when available.
// Also atomically accumulates p/NHEAD into attnW[b, s, t] (head mean).
__global__ __launch_bounds__(128)
void attn_wmma(const _Float16* __restrict__ Q, const _Float16* __restrict__ Kd,
               const _Float16* __restrict__ Vd, _Float16* __restrict__ ctx,
               float* __restrict__ attnW) {
    constexpr int LD = 72;                           // 144B rows, 16B aligned
    // TDM padding codes matching LD=72: 64 halves = 32 DWORDs data (code 4),
    // 8 halves = 4 DWORDs pad (code 3).
    __shared__ _Float16 Qs [64 * LD];
    __shared__ _Float16 Ks [64 * LD];
    __shared__ _Float16 Vts[64 * LD];                // transposed: [hd][t]
    __shared__ _Float16 Ps [4][16 * LD];             // per-wave p tile

    const int t    = threadIdx.x;
    const int lane = t & 31;
    const int wave = t >> 5;
    const int qt = blockIdx.x, h = blockIdx.y, b = blockIdx.z;
    const int qbase = qt * 64;
    const size_t baseBH = (size_t)b * SEQ * EMBED + (size_t)h * HDIM;

    // ---- load Q tile 64x64 (f16) once ----
#if USE_TDM
    if (t < 32) {
        tdm_load_2d_f16((unsigned int)(uintptr_t)&Qs[0],
                        &Q[baseBH + (size_t)qbase * EMBED],
                        64, 64, EMBED, 1u, 4u, 3u);
        wait_tensor0();
    }
#else
#pragma unroll
    for (int i = 0; i < 4; ++i) {
        int idx = t + i * 128;                       // 512 chunks of 8 halves
        int row = idx >> 3, c8 = idx & 7;
        *(v8h*)(&Qs[row * LD + c8 * 8]) =
            *(const v8h*)(&Q[baseBH + (size_t)(qbase + row) * EMBED + c8 * 8]);
    }
#endif
    __syncthreads();

    const int half = lane >> 4, col0 = lane & 15;
    const int row_local = wave * 16 + half * 8;      // + r

    v16h qf[2];
    qf[0] = frag_a(Qs, LD, wave * 16, 0);
    qf[1] = frag_a(Qs, LD, wave * 16, 32);

    float m[8], l[8];
#pragma unroll
    for (int r = 0; r < 8; ++r) { m[r] = -3.0e38f; l[r] = 0.0f; }

    // ================= pass 1: row max + sum-exp =================
    for (int kt = 0; kt <= qt; ++kt) {
        const int tb = kt * 64;
#if USE_TDM
        if (t < 32) {
            tdm_load_2d_f16((unsigned int)(uintptr_t)&Ks[0],
                            &Kd[baseBH + (size_t)tb * EMBED],
                            64, 64, EMBED, 1u, 4u, 3u);
            wait_tensor0();
        }
#else
#pragma unroll
        for (int i = 0; i < 4; ++i) {
            int idx = t + i * 128;
            int row = idx >> 3, c8 = idx & 7;
            *(v8h*)(&Ks[row * LD + c8 * 8]) =
                *(const v8h*)(&Kd[baseBH + (size_t)(tb + row) * EMBED + c8 * 8]);
        }
#endif
        if (kt < qt)   // prefetch next K tile into L2 while computing
            __builtin_prefetch(&Kd[baseBH + (size_t)(tb + 64 + (t & 63)) * EMBED], 0, 0);
        __syncthreads();

        v8f d[4];
#pragma unroll
        for (int nt = 0; nt < 4; ++nt) {
            v8f c = {};
            c = __builtin_amdgcn_wmma_f32_16x16x32_f16(false, qf[0], false,
                    frag_b(Ks, LD, nt * 16, 0), (short)0, c, false, false);
            c = __builtin_amdgcn_wmma_f32_16x16x32_f16(false, qf[1], false,
                    frag_b(Ks, LD, nt * 16, 32), (short)0, c, false, false);
            d[nt] = c;
        }
#pragma unroll
        for (int r = 0; r < 8; ++r) {
            const int row_g = qbase + row_local + r;
            float sc[4], mx = -3.0e38f;
#pragma unroll
            for (int nt = 0; nt < 4; ++nt) {
                int col_g = tb + nt * 16 + col0;
                float s = d[nt][r] * SCALE_F;
                if (col_g > row_g) s = -3.0e38f;
                sc[nt] = s;
                mx = fmaxf(mx, s);
            }
#pragma unroll
            for (int x = 1; x < 16; x <<= 1) mx = fmaxf(mx, __shfl_xor(mx, x, 32));
            float mnew = fmaxf(m[r], mx);
            float se = 0.0f;
#pragma unroll
            for (int nt = 0; nt < 4; ++nt) se += __expf(sc[nt] - mnew);
#pragma unroll
            for (int x = 1; x < 16; x <<= 1) se += __shfl_xor(se, x, 32);
            l[r] = l[r] * __expf(m[r] - mnew) + se;
            m[r] = mnew;
        }
        __syncthreads();
    }

    float rl[8];
#pragma unroll
    for (int r = 0; r < 8; ++r) rl[r] = 1.0f / l[r];

    // ================= pass 2: probs -> attnW atomics + P@V =================
    v8f acc[4] = {};
    _Float16* Pw = &Ps[wave][0];

    for (int kt = 0; kt <= qt; ++kt) {
        const int tb = kt * 64;
#if USE_TDM
        if (t < 32)
            tdm_load_2d_f16((unsigned int)(uintptr_t)&Ks[0],
                            &Kd[baseBH + (size_t)tb * EMBED],
                            64, 64, EMBED, 1u, 4u, 3u);
#else
#pragma unroll
        for (int i = 0; i < 4; ++i) {
            int idx = t + i * 128;
            int row = idx >> 3, c8 = idx & 7;
            *(v8h*)(&Ks[row * LD + c8 * 8]) =
                *(const v8h*)(&Kd[baseBH + (size_t)(tb + row) * EMBED + c8 * 8]);
        }
#endif
        // V tile, transposed into [hd][t] (overlaps with TDM in flight)
#pragma unroll
        for (int i = 0; i < 32; ++i) {
            int idx = t + i * 128;                   // 4096 elements
            int tr = idx >> 6, hd = idx & 63;
            Vts[hd * LD + tr] = Vd[baseBH + (size_t)(tb + tr) * EMBED + hd];
        }
#if USE_TDM
        if (t < 32) wait_tensor0();
#endif
        __syncthreads();

        v8f d[4];
#pragma unroll
        for (int nt = 0; nt < 4; ++nt) {
            v8f c = {};
            c = __builtin_amdgcn_wmma_f32_16x16x32_f16(false, qf[0], false,
                    frag_b(Ks, LD, nt * 16, 0), (short)0, c, false, false);
            c = __builtin_amdgcn_wmma_f32_16x16x32_f16(false, qf[1], false,
                    frag_b(Ks, LD, nt * 16, 32), (short)0, c, false, false);
            d[nt] = c;
        }
#pragma unroll
        for (int nt = 0; nt < 4; ++nt) {
            const int col_g = tb + nt * 16 + col0;
#pragma unroll
            for (int r = 0; r < 8; ++r) {
                const int row_g = qbase + row_local + r;
                float p = 0.0f;
                if (col_g <= row_g) {
                    p = __expf(d[nt][r] * SCALE_F - m[r]) * rl[r];
                    atomicAdd(&attnW[((size_t)b * SEQ + row_g) * SEQ + col_g],
                              p * (1.0f / NHEAD));
                }
                Pw[(half * 8 + r) * LD + nt * 16 + col0] = (_Float16)p;
            }
        }
        // P (16x64) @ V (64x64): re-gather p from LDS in A-fragment layout
        v16h pf0 = frag_a(Pw, LD, 0, 0);
        v16h pf1 = frag_a(Pw, LD, 0, 32);
#pragma unroll
        for (int nt2 = 0; nt2 < 4; ++nt2) {
            acc[nt2] = __builtin_amdgcn_wmma_f32_16x16x32_f16(false, pf0, false,
                           frag_b(Vts, LD, nt2 * 16, 0), (short)0, acc[nt2], false, false);
            acc[nt2] = __builtin_amdgcn_wmma_f32_16x16x32_f16(false, pf1, false,
                           frag_b(Vts, LD, nt2 * 16, 32), (short)0, acc[nt2], false, false);
        }
        __syncthreads();
    }

    // ---- write context (f16) in [B, S, D] with head offset ----
#pragma unroll
    for (int nt2 = 0; nt2 < 4; ++nt2)
#pragma unroll
        for (int r = 0; r < 8; ++r) {
            int row_g = qbase + row_local + r;
            int c = h * HDIM + nt2 * 16 + col0;
            ctx[((size_t)b * SEQ + row_g) * EMBED + c] = (_Float16)acc[nt2][r];
        }
}

// ---------------------------------------------------------------------------
extern "C" void kernel_launch(void* const* d_in, const int* in_sizes, int n_in,
                              void* d_out, int out_size, void* d_ws, size_t ws_size,
                              hipStream_t stream) {
    const float* query = (const float*)d_in[0];
    const float* key   = (const float*)d_in[1];
    const float* value = (const float*)d_in[2];
    const float* Wq = (const float*)d_in[3];
    const float* bq = (const float*)d_in[4];
    const float* Wk = (const float*)d_in[5];
    const float* bk = (const float*)d_in[6];
    const float* Wv = (const float*)d_in[7];
    const float* bv = (const float*)d_in[8];
    const float* Wo = (const float*)d_in[9];
    const float* bo = (const float*)d_in[10];

    float* out   = (float*)d_out;                       // [B,S,D]
    float* attnW = out + (size_t)BATCH * SEQ * EMBED;   // [B,S,S]

    // f16 workspace: q, k, v, ctx  (each B*S*D halves)
    const size_t per = (size_t)BATCH * SEQ * EMBED;
    _Float16* qws   = (_Float16*)d_ws;
    _Float16* kws   = qws + per;
    _Float16* vws   = kws + per;
    _Float16* ctxws = vws + per;

    // attn weights are accumulated atomically -> zero them each call
    (void)hipMemsetAsync(attnW, 0, (size_t)BATCH * SEQ * SEQ * sizeof(float), stream);

    dim3 gproj(MROWS / 128, EMBED / 64);
    gemm_bias_wmma<float, _Float16><<<gproj, 256, 0, stream>>>(query, Wq, bq, qws, MROWS, EMBED, EMBED);
    gemm_bias_wmma<float, _Float16><<<gproj, 256, 0, stream>>>(key,   Wk, bk, kws, MROWS, EMBED, EMBED);
    gemm_bias_wmma<float, _Float16><<<gproj, 256, 0, stream>>>(value, Wv, bv, vws, MROWS, EMBED, EMBED);

    attn_wmma<<<dim3(SEQ / 64, NHEAD, BATCH), 128, 0, stream>>>(qws, kws, vws, ctxws, attnW);

    gemm_bias_wmma<_Float16, float><<<gproj, 256, 0, stream>>>(ctxws, Wo, bo, out, MROWS, EMBED, EMBED);
}
